// LorentzSelfAttention_57767310131243
// MI455X (gfx1250) — matrix operand order repaired and checked by reading, not compiled, not run on hardware
//
#include <hip/hip_runtime.h>
#include <hip/hip_bf16.h>
#include <stdint.h>

typedef __attribute__((ext_vector_type(16))) _Float16 v16h;
typedef __attribute__((ext_vector_type(8)))  _Float16 v8h;
typedef __attribute__((ext_vector_type(4)))  _Float16 v4h;
typedef __attribute__((ext_vector_type(8)))  float    v8f;
typedef __attribute__((ext_vector_type(4)))  float    v4f;
typedef __attribute__((ext_vector_type(4)))  int      v4i;

#define BB 4
#define LL 4096
#define DD 64
#define BM 64          // query rows per workgroup (4 waves x 16)
#define BN 64          // keys per KV tile
#define KPITCH 72      // LDS row pitch in halves (144 B; 16B-aligned subchunks)
#define NEG_BIG (-4294967295.0f)

// ---- optional CDNA5 async-to-LDS path (guarded; falls back cleanly) ----
#if defined(__has_builtin)
#  if __has_builtin(__builtin_amdgcn_global_load_async_to_lds_b128) && \
      __has_builtin(__builtin_amdgcn_s_wait_asynccnt)
#    define USE_ASYNC 1
#  endif
#endif
#ifndef USE_ASYNC
#  define USE_ASYNC 0
#endif

#if USE_ASYNC
__device__ __forceinline__ void async_cp16(void* lds_dst, const void* gsrc) {
  // Builtin signature (from clang diagnostic): (int4 AS(1)*, int4 AS(3)*, Ii, Ii).
  // Generic LDS pointers keep the byte offset in the low 32 bits (aperture in
  // the high bits), so the truncating inttoptr yields the AS(3) address.
  __attribute__((address_space(1))) v4i* g =
      (__attribute__((address_space(1))) v4i*)(uintptr_t)gsrc;
  __attribute__((address_space(3))) v4i* l =
      (__attribute__((address_space(3))) v4i*)(uintptr_t)(uint32_t)(uintptr_t)lds_dst;
  __builtin_amdgcn_global_load_async_to_lds_b128(g, l, 0, 0);
}
#endif

__device__ __forceinline__ v16h join16(v8h lo, v8h hi) {
  v16h r;
#pragma unroll
  for (int i = 0; i < 8; ++i) { r[i] = lo[i]; r[i + 8] = hi[i]; }
  return r;
}

__device__ __forceinline__ v8f wmma16(v16h a, v16h b, v8f c) {
  // D = A(16x32 f16) * B(32x16 f16) + C(16x16 f32)
  return __builtin_amdgcn_wmma_f32_16x16x32_f16(
      /*neg_a=*/false, a, /*neg_b=*/false, b,
      /*c_mod=*/(short)0, c, /*reuse_a=*/false, /*reuse_b=*/false);
}

// ---- pass 1 (async path only): Q f32 -> f16, row-major + transposed ----
__global__ __launch_bounds__(256)
void cvt_q_kernel(const float* __restrict__ Q,
                  _Float16* __restrict__ Qh,      // [B][L][D]
                  _Float16* __restrict__ QhT) {   // [B][D][L]
  int g = (blockIdx.x * 256 + threadIdx.x) * 4;   // grid covers B*L*D
  int b = g >> 18;                                 // / (L*D)
  int r = (g >> 6) & (LL - 1);
  int c = g & (DD - 1);
  v4f q4 = *(const v4f*)&Q[g];
  v4h h4;
#pragma unroll
  for (int j = 0; j < 4; ++j) h4[j] = (_Float16)q4[j];
  *(v4h*)&Qh[g] = h4;
#pragma unroll
  for (int j = 0; j < 4; ++j)
    QhT[(((size_t)b * DD) + c + j) * LL + r] = h4[j];
}

// ---- flash-attention kernel; ASYNC selects the double-buffered async path ----
template <bool ASYNC>
__global__ __launch_bounds__(128)
void lorentz_attn_fa_kernel(const float* __restrict__ Q,
                            const unsigned char* __restrict__ maskp,
                            const float* __restrict__ scale_p,
                            const float* __restrict__ bias_p,
                            float* __restrict__ out,
                            const _Float16* __restrict__ Qh,
                            const _Float16* __restrict__ QhT) {
  constexpr int NBUF = ASYNC ? 2 : 1;
  __shared__ _Float16 kvR[NBUF][BN][KPITCH];      // [key][d]
  __shared__ _Float16 kvT[NBUF][DD][KPITCH];      // [d][key]
  __shared__ _Float16 pbuf[4][16][KPITCH];        // per-wave P strip [m][key]

  const int tid  = threadIdx.x;
  const int wave = tid >> 5;
  const int lane = tid & 31;
  const int half = lane >> 4;
  const int lan  = lane & 15;

  const int nqb = LL / BM;
  const int b   = blockIdx.x / nqb;
  const int q0  = (blockIdx.x % nqb) * BM;

  const float LOG2E = 1.4426950408889634f;
  const float scale = scale_p[0];
  const float bias  = bias_p[0];
  const float alpha = (2.0f / scale) * LOG2E;          // folds into q (exp2 domain)
  const float beta  = (2.0f / scale + bias) * LOG2E;   // additive (softmax-invariant)

  const float* Qb = Q + (size_t)b * LL * DD;
  const unsigned char* mb = maskp + (size_t)b * LL;

  // ---- this wave's Q A-fragments (16x32 f16 x2), signs & alpha folded in ----
  const int qrow = q0 + wave * 16 + lan;
  const int kb   = half * 8;
  v16h qa[2];
#pragma unroll
  for (int s = 0; s < 2; ++s) {
#pragma unroll
    for (int h = 0; h < 16; ++h) {
      int d = s * 32 + half * 16 + h;
      float x = Qb[qrow * DD + d] * alpha;
      if (d == 0) x = -x;                  // Minkowski sign on time component
      qa[s][h] = (_Float16)x;
    }
  }

  // ---- flash state (row m = v + 8*half; 16 lanes share each row) ----
  float mrow[8], lrow[8];
  v8f oacc[4];
#pragma unroll
  for (int v = 0; v < 8; ++v) { mrow[v] = -3.0e38f; lrow[v] = 0.0f; }
#pragma unroll
  for (int t = 0; t < 4; ++t)
#pragma unroll
    for (int v = 0; v < 8; ++v) oacc[t][v] = 0.0f;

#if USE_ASYNC
  const _Float16* QhR = Qh  + (size_t)b * LL * DD;   // [key][d]
  const _Float16* QhC = QhT + (size_t)b * DD * LL;   // [d][key]
  auto issue_tile = [&](int j0, int buf) {
    if constexpr (ASYNC) {
#pragma unroll
      for (int c = 0; c < 4; ++c) {
        int k    = c * 128 + tid;        // 512 x 16B chunks per tile copy
        int row  = k >> 3;
        int col8 = (k & 7) * 8;
        async_cp16(&kvR[buf][row][col8], QhR + (size_t)(j0 + row) * DD + col8);
        async_cp16(&kvT[buf][row][col8], QhC + (size_t)row * LL + j0 + col8);
      }
    }
  };
  if constexpr (ASYNC) issue_tile(0, 0);
#endif

  int buf = 0;
  for (int j0 = 0; j0 <= q0; j0 += BN) {
    if constexpr (ASYNC) {
#if USE_ASYNC
      __builtin_amdgcn_s_wait_asynccnt(0);   // my chunks of tile j0 landed
      __syncthreads();                       // everyone's chunks landed; prev buf free
      if (j0 + BN <= q0) issue_tile(j0 + BN, buf ^ 1);   // prefetch next tile
#endif
    } else {
      __syncthreads();
      // cooperative tile load: f32 -> f16, row-major AND transposed
#pragma unroll
      for (int it = 0; it < 8; ++it) {
        int i = (it * 128 + tid) * 4;
        int r = i >> 6, c = i & 63;
        v4f q4 = *(const v4f*)&Qb[(j0 + r) * DD + c];
        v4h h4;
#pragma unroll
        for (int j = 0; j < 4; ++j) h4[j] = (_Float16)q4[j];
        *(v4h*)&kvR[0][r][c] = h4;
#pragma unroll
        for (int j = 0; j < 4; ++j) kvT[0][c + j][r] = h4[j];
      }
      __syncthreads();
      if (j0 + BN <= q0)
        __builtin_prefetch(&Qb[(size_t)(j0 + BN) * DD] + tid * 32, 0, 1);
    }

    // ---- S = (q*signs*alpha) @ K^T (+beta), masked, 16x64 strip per wave ----
    float p[4][8];
#pragma unroll
    for (int nt = 0; nt < 4; ++nt) {
      const int key = nt * 16 + lan;   // B layout: lanes 0-15 K=0..15, 16-31 K=16..31
      v16h b0 = join16(*(const v8h*)&kvR[buf][key][half * 16],
                       *(const v8h*)&kvR[buf][key][half * 16 + 8]);
      v16h b1 = join16(*(const v8h*)&kvR[buf][key][32 + half * 16],
                       *(const v8h*)&kvR[buf][key][32 + half * 16 + 8]);
      v8f acc = {};
      acc = wmma16(qa[0], b0, acc);
      acc = wmma16(qa[1], b1, acc);

      const int kj = j0 + nt * 16 + lan;
      const bool pm = mb[kj] != 0;
#pragma unroll
      for (int v = 0; v < 8; ++v) {
        int qi = q0 + wave * 16 + v + half * 8;
        float sv = acc[v] + beta;
        if (pm || kj > qi) sv = NEG_BIG * LOG2E;   // finite, matches reference
        p[nt][v] = sv;
      }
    }

    // ---- online softmax update (exp2 domain) ----
#pragma unroll
    for (int v = 0; v < 8; ++v) {
      float mx = fmaxf(fmaxf(p[0][v], p[1][v]), fmaxf(p[2][v], p[3][v]));
#pragma unroll
      for (int o = 1; o < 16; o <<= 1) mx = fmaxf(mx, __shfl_xor(mx, o, 32));
      float mnew = fmaxf(mrow[v], mx);
      float corr = exp2f(mrow[v] - mnew);
      float rs = 0.0f;
#pragma unroll
      for (int nt = 0; nt < 4; ++nt) {
        float e = exp2f(p[nt][v] - mnew);
        p[nt][v] = e;
        rs += e;
      }
#pragma unroll
      for (int o = 1; o < 16; o <<= 1) rs += __shfl_xor(rs, o, 32);
      lrow[v] = lrow[v] * corr + rs;
      mrow[v] = mnew;
#pragma unroll
      for (int t = 0; t < 4; ++t) oacc[t][v] *= corr;
    }

    // ---- C-layout P -> A-layout via per-wave LDS strip ----
#pragma unroll
    for (int nt = 0; nt < 4; ++nt)
#pragma unroll
      for (int v = 0; v < 8; ++v)
        pbuf[wave][v + half * 8][nt * 16 + lan] = (_Float16)p[nt][v];

    v16h pa[2];
#pragma unroll
    for (int s = 0; s < 2; ++s)
      pa[s] = join16(*(const v8h*)&pbuf[wave][lan][s * 32 + half * 16],
                     *(const v8h*)&pbuf[wave][lan][s * 32 + half * 16 + 8]);

    // ---- O += P @ V  (V columns contiguous in kvT rows) ----
#pragma unroll
    for (int dt = 0; dt < 4; ++dt) {
      const int dcol = dt * 16 + lan;
      v16h b0 = join16(*(const v8h*)&kvT[buf][dcol][half * 16],
                       *(const v8h*)&kvT[buf][dcol][half * 16 + 8]);
      v16h b1 = join16(*(const v8h*)&kvT[buf][dcol][32 + half * 16],
                       *(const v8h*)&kvT[buf][dcol][32 + half * 16 + 8]);
      oacc[dt] = wmma16(pa[0], b0, oacc[dt]);
      oacc[dt] = wmma16(pa[1], b1, oacc[dt]);
    }
    if constexpr (ASYNC) buf ^= 1;
  }

  // ---- epilogue: mu = O/l ; Lorentz norm ; out = mu / sqrt(max(|ln|,eps)) ----
#pragma unroll
  for (int v = 0; v < 8; ++v) {
    float linv = 1.0f / lrow[v];
    float muv[4];
    float ss = 0.0f;
#pragma unroll
    for (int dt = 0; dt < 4; ++dt) {
      float mu = oacc[dt][v] * linv;
      muv[dt] = mu;
      float sq = mu * mu;
      if (dt == 0 && lan == 0) sq = -sq;   // d==0 time component
      ss += sq;
    }
#pragma unroll
    for (int o = 1; o < 16; o <<= 1) ss += __shfl_xor(ss, o, 32);
    float inv = 1.0f / sqrtf(fmaxf(fabsf(ss), 1e-8f));
    int row = q0 + wave * 16 + v + half * 8;
#pragma unroll
    for (int dt = 0; dt < 4; ++dt)
      out[((size_t)b * LL + row) * DD + dt * 16 + lan] = muv[dt] * inv;
  }
}

extern "C" void kernel_launch(void* const* d_in, const int* in_sizes, int n_in,
                              void* d_out, int out_size, void* d_ws, size_t ws_size,
                              hipStream_t stream) {
  const float*         Q     = (const float*)d_in[0];
  const unsigned char* maskp = (const unsigned char*)d_in[1];
  const float*         scale = (const float*)d_in[2];
  const float*         bias  = (const float*)d_in[3];
  float*               out   = (float*)d_out;

  dim3 grid(BB * (LL / BM));   // 4 batches x 64 query blocks = 256 WGs
  dim3 block(128);             // 4 waves (wave32)

#if USE_ASYNC
  const size_t halfbytes = (size_t)BB * LL * DD * sizeof(_Float16);  // 2 MB
  if (ws_size >= 2 * halfbytes) {
    _Float16* Qh  = (_Float16*)d_ws;
    _Float16* QhT = (_Float16*)((char*)d_ws + halfbytes);
    hipLaunchKernelGGL(cvt_q_kernel, dim3((BB * LL * DD) / (256 * 4)), dim3(256),
                       0, stream, Q, Qh, QhT);
    hipLaunchKernelGGL((lorentz_attn_fa_kernel<true>), grid, block, 0, stream,
                       Q, maskp, scale, bias, out, Qh, QhT);
    return;
  }
#endif
  hipLaunchKernelGGL((lorentz_attn_fa_kernel<false>), grid, block, 0, stream,
                     Q, maskp, scale, bias, out, nullptr, nullptr);
}
